// RNN_91182155694521
// MI455X (gfx1250) — compile-verified, matrix operand order checked by reading, not run on hardware
//
#include <hip/hip_runtime.h>

// ---------------- problem constants ----------------
#define T_STEPS 1024
#define BATCH   64
#define IDIM    512
#define HDIM    512
#define KDIM    (IDIM + HDIM)      // 1024 : fused [x | h] K dimension
#define NWG     16                 // persistent workgroups
#define SLAB    (HDIM / NWG)       // 32 hidden columns per WG
#define GCOLS   (3 * SLAB)         // 96 gate columns per WG (r,z,n)
#define NTHREADS 256               // 8 wave32

typedef __attribute__((ext_vector_type(16))) __bf16    v16bf;
typedef __attribute__((ext_vector_type(8)))  float     v8f;
typedef __attribute__((ext_vector_type(4)))  unsigned  u32x4;
typedef __attribute__((ext_vector_type(4)))  float     f32x4;

union Frag {            // one WMMA 16-bit operand fragment: 8 VGPRs / lane
  u32x4 q[2];
  v16bf v;
};

static __device__ __forceinline__ unsigned short f2bf(float f) {
  unsigned u = __float_as_uint(f);
  u += 0x7FFFu + ((u >> 16) & 1u);          // round-to-nearest-even
  return (unsigned short)(u >> 16);
}
static __device__ __forceinline__ float bf2f(unsigned short h) {
  return __uint_as_float(((unsigned)h) << 16);
}

// --------- prep: split weights to bf16 hi/lo, zero h0 & barrier ----------
__global__ void gru_prep(const float* __restrict__ W_ih, const float* __restrict__ W_hh,
                         unsigned short* __restrict__ Whi, unsigned short* __restrict__ Wlo,
                         float* __restrict__ h0, unsigned* __restrict__ ctr) {
  int gid = blockIdx.x * blockDim.x + threadIdx.x;
  if (gid < 3 * HDIM * KDIM) {
    int r = gid / KDIM, k = gid % KDIM;
    float w = (k < IDIM) ? W_ih[r * IDIM + k] : W_hh[r * HDIM + (k - IDIM)];
    unsigned short hi = f2bf(w);
    Whi[gid] = hi;
    Wlo[gid] = f2bf(w - bf2f(hi));
  }
  if (gid < BATCH * HDIM) h0[gid] = 0.f;
  if (gid == 0) *ctr = 0u;
}

// --------- persistent GRU recurrence ----------
#define WMMA_BF16(A, B, C) \
  __builtin_amdgcn_wmma_f32_16x16x32_bf16(false, (A), false, (B), (short)0, (C), false, false)

__global__ void __launch_bounds__(NTHREADS, 1)
gru_persistent(const float* __restrict__ x, const float* __restrict__ pads,
               const unsigned short* __restrict__ Whi, const unsigned short* __restrict__ Wlo,
               const float* __restrict__ b_ih, const float* __restrict__ b_hh,
               float* __restrict__ hbuf0, float* __restrict__ hbuf1,
               unsigned* __restrict__ ctr, float* __restrict__ out) {
  extern __shared__ char smem[];
  unsigned short* Ahi = (unsigned short*)smem;          // [64][1024] bf16 hi   (128 KB)
  unsigned short* Alo = Ahi + BATCH * KDIM;             // [64][1024] bf16 lo   (128 KB)
  float* Sx = (float*)smem;                             // alias: x-part sums [64][96]
  float* Sh = Sx + BATCH * GCOLS;                       // alias: h-part sums [64][96]

  const int tid  = threadIdx.x;
  const int wg   = blockIdx.x;
  const int wave = tid >> 5, lane = tid & 31;
  const int rb     = wave & 3;          // 16-row batch block
  const int cbBase = (wave >> 2) * 3;   // 3 of 6 gate-column blocks
  const int c0     = wg * SLAB;

  const int Mrow  = rb * 16 + (lane & 15);
  const int khalf = (lane >= 16) ? 8 : 0;

  // loop-invariant fragment base pointers
  const unsigned short* aBaseHi = Ahi + Mrow * KDIM + khalf;
  const unsigned short* aBaseLo = Alo + Mrow * KDIM + khalf;
  const unsigned short* wBaseHi[3];
  const unsigned short* wBaseLo[3];
#pragma unroll
  for (int i = 0; i < 3; ++i) {
    int cb = cbBase + i;
    int wrow = (cb >> 1) * HDIM + c0 + (cb & 1) * 16 + (lane & 15);
    wBaseHi[i] = Whi + (size_t)wrow * KDIM + khalf;
    wBaseLo[i] = Wlo + (size_t)wrow * KDIM + khalf;
  }

  const float* hcur  = hbuf0;
  float*       hnext = hbuf1;

#pragma unroll 1
  for (int t = 0; t < T_STEPS; ++t) {
    // ---- stage A = [x_t | h] into LDS as bf16 hi/lo, 8 elements per iteration ----
    // 2x global_load_b128 -> split -> 2x ds_store_b128 (hi, lo)
    const float* xt = x + (size_t)t * BATCH * IDIM;
#pragma unroll 1
    for (int e8 = tid; e8 < (BATCH * KDIM) / 8; e8 += NTHREADS) {
      int b = e8 >> 7, k0 = (e8 & 127) * 8;            // KDIM/8 = 128 chunks per row
      const float* src;
      if (k0 < IDIM) {
        src = xt + b * IDIM + k0;
        if (t + 1 < T_STEPS) __builtin_prefetch(src + BATCH * IDIM, 0, 1);
      } else {
        src = hcur + b * HDIM + (k0 - IDIM);
      }
      f32x4 f0 = *(const f32x4*)(src);
      f32x4 f1 = *(const f32x4*)(src + 4);
      float f[8] = {f0[0], f0[1], f0[2], f0[3], f1[0], f1[1], f1[2], f1[3]};
      u32x4 vh, vl;
#pragma unroll
      for (int i = 0; i < 4; ++i) {
        unsigned short h0 = f2bf(f[2 * i]);
        unsigned short h1 = f2bf(f[2 * i + 1]);
        unsigned short l0 = f2bf(f[2 * i]     - bf2f(h0));
        unsigned short l1 = f2bf(f[2 * i + 1] - bf2f(h1));
        vh[i] = (unsigned)h0 | ((unsigned)h1 << 16);
        vl[i] = (unsigned)l0 | ((unsigned)l1 << 16);
      }
      *(u32x4*)(Ahi + e8 * 8) = vh;
      *(u32x4*)(Alo + e8 * 8) = vl;
    }
    __syncthreads();

    // ---- split-bf16 WMMA GEMM ----
    // x-part (K 0..511) and h-part (K 512..1023) kept in separate accumulators:
    // the n-gate needs  tanh(i_n + r * h_n)  so the two halves cannot be summed.
    v8f ax0 = {}, ax1 = {}, ax2 = {}, ah0 = {}, ah1 = {}, ah2 = {};

#pragma unroll 2
    for (int ks = 0; ks < KDIM / 64; ++ks) {          // x contribution
      const int kb = ks * 32;
      Frag Af, Al_;
      Af.q[0]  = *(const u32x4*)(aBaseHi + kb);
      Af.q[1]  = *(const u32x4*)(aBaseHi + kb + 16);
      Al_.q[0] = *(const u32x4*)(aBaseLo + kb);
      Al_.q[1] = *(const u32x4*)(aBaseLo + kb + 16);
#pragma unroll
      for (int i = 0; i < 3; ++i) {
        Frag Bh, Bl;
        Bh.q[0] = *(const u32x4*)(wBaseHi[i] + kb);
        Bh.q[1] = *(const u32x4*)(wBaseHi[i] + kb + 16);
        Bl.q[0] = *(const u32x4*)(wBaseLo[i] + kb);
        Bl.q[1] = *(const u32x4*)(wBaseLo[i] + kb + 16);
        v8f& acc = (i == 0) ? ax0 : (i == 1) ? ax1 : ax2;
        acc = WMMA_BF16(Af.v,  Bh.v, acc);
        acc = WMMA_BF16(Af.v,  Bl.v, acc);
        acc = WMMA_BF16(Al_.v, Bh.v, acc);
      }
    }

#pragma unroll 2
    for (int ks = KDIM / 64; ks < KDIM / 32; ++ks) {  // h contribution
      const int kb = ks * 32;
      Frag Af, Al_;
      Af.q[0]  = *(const u32x4*)(aBaseHi + kb);
      Af.q[1]  = *(const u32x4*)(aBaseHi + kb + 16);
      Al_.q[0] = *(const u32x4*)(aBaseLo + kb);
      Al_.q[1] = *(const u32x4*)(aBaseLo + kb + 16);
#pragma unroll
      for (int i = 0; i < 3; ++i) {
        Frag Bh, Bl;
        Bh.q[0] = *(const u32x4*)(wBaseHi[i] + kb);
        Bh.q[1] = *(const u32x4*)(wBaseHi[i] + kb + 16);
        Bl.q[0] = *(const u32x4*)(wBaseLo[i] + kb);
        Bl.q[1] = *(const u32x4*)(wBaseLo[i] + kb + 16);
        v8f& acc = (i == 0) ? ah0 : (i == 1) ? ah1 : ah2;
        acc = WMMA_BF16(Af.v,  Bh.v, acc);
        acc = WMMA_BF16(Af.v,  Bl.v, acc);
        acc = WMMA_BF16(Al_.v, Bh.v, acc);
      }
    }
    __syncthreads();   // everyone done reading A before S aliases over it

    // ---- spill pre-activation tiles to LDS (C/D layout: row = rb*16 + v + hi-half*8) ----
    {
      int srow0 = rb * 16 + ((lane >= 16) ? 8 : 0);
#pragma unroll
      for (int i = 0; i < 3; ++i) {
        int j = (cbBase + i) * 16 + (lane & 15);
        v8f sx = (i == 0) ? ax0 : ((i == 1) ? ax1 : ax2);
        v8f sh = (i == 0) ? ah0 : ((i == 1) ? ah1 : ah2);
#pragma unroll
        for (int v = 0; v < 8; ++v) {
          Sx[(srow0 + v) * GCOLS + j] = sx[v];
          Sh[(srow0 + v) * GCOLS + j] = sh[v];
        }
      }
    }
    __syncthreads();

    // ---- elementwise gates + padding carry; write h_next and both outputs ----
#pragma unroll 1
    for (int e = tid; e < BATCH * SLAB; e += NTHREADS) {
      int b = e >> 5, cl = e & (SLAB - 1);
      int c = c0 + cl;
      float pre_r = Sx[b * GCOLS + cl] + Sh[b * GCOLS + cl] + b_ih[c] + b_hh[c];
      float pre_z = Sx[b * GCOLS + SLAB + cl] + Sh[b * GCOLS + SLAB + cl]
                  + b_ih[HDIM + c] + b_hh[HDIM + c];
      float gin = Sx[b * GCOLS + 2 * SLAB + cl] + b_ih[2 * HDIM + c];
      float ghn = Sh[b * GCOLS + 2 * SLAB + cl] + b_hh[2 * HDIM + c];
      float r = 1.f / (1.f + expf(-pre_r));
      float z = 1.f / (1.f + expf(-pre_z));
      float n = tanhf(gin + r * ghn);
      float hold = hcur[b * HDIM + c];
      float hc = (1.f - z) * n + z * hold;
      float p  = pads[t * BATCH + b];
      float hn = p * hold + (1.f - p) * hc;
      hnext[b * HDIM + c] = hn;
      size_t oidx = (size_t)b * (T_STEPS * HDIM) + (size_t)t * HDIM + c;
      out[oidx] = hn;                                        // outputs [B,T,H]
      out[(size_t)BATCH * T_STEPS * HDIM + oidx] = hn;       // states  [B,T,H]
    }

    // ---- one grid barrier per timestep (monotonic counter, release/acquire) ----
    __syncthreads();
    if (tid == 0) {
      __threadfence();
      __hip_atomic_fetch_add(ctr, 1u, __ATOMIC_RELEASE, __HIP_MEMORY_SCOPE_AGENT);
      unsigned tgt = (unsigned)NWG * (unsigned)(t + 1);
      while (__hip_atomic_load(ctr, __ATOMIC_ACQUIRE, __HIP_MEMORY_SCOPE_AGENT) < tgt)
        __builtin_amdgcn_s_sleep(2);
    }
    __syncthreads();
    __threadfence();   // acquire: make peers' h_next visible to all waves

    const float* tmp = hcur; hcur = hnext; hnext = (float*)tmp;
  }
}

extern "C" void kernel_launch(void* const* d_in, const int* in_sizes, int n_in,
                              void* d_out, int out_size, void* d_ws, size_t ws_size,
                              hipStream_t stream) {
  (void)in_sizes; (void)n_in; (void)out_size; (void)ws_size;
  const float* x    = (const float*)d_in[0];
  const float* pads = (const float*)d_in[1];
  const float* W_ih = (const float*)d_in[2];
  const float* W_hh = (const float*)d_in[3];
  const float* b_ih = (const float*)d_in[4];
  const float* b_hh = (const float*)d_in[5];

  char* ws = (char*)d_ws;
  unsigned short* Whi = (unsigned short*)ws;                          // 3 MB
  unsigned short* Wlo = Whi + (size_t)3 * HDIM * KDIM;                // 3 MB
  float* h0 = (float*)(Wlo + (size_t)3 * HDIM * KDIM);                // 128 KB
  float* h1 = h0 + BATCH * HDIM;                                      // 128 KB
  unsigned* ctr = (unsigned*)(h1 + BATCH * HDIM);

  int prepN = 3 * HDIM * KDIM;
  gru_prep<<<(prepN + 255) / 256, 256, 0, stream>>>(W_ih, W_hh, Whi, Wlo, h0, ctr);

  size_t lds = (size_t)2 * BATCH * KDIM * sizeof(unsigned short);     // 256 KB / WGP
  gru_persistent<<<NWG, NTHREADS, lds, stream>>>(x, pads, Whi, Wlo, b_ih, b_hh,
                                                 h0, h1, ctr, (float*)d_out);
}
